// GraphSAGE_22411139350716
// MI455X (gfx1250) — compile-verified, matrix operand order checked by reading, not run on hardware
//
#include <hip/hip_runtime.h>

// ---------------------------------------------------------------------------
// GraphSAGE pipeline for MI455X (gfx1250, wave32, WMMA).
// All GEMMs run through v_wmma_f32_16x16x32_bf16 (bf16 in, f32 accum).
// Weights are pre-transposed to [N,K] in the prep kernels so both A and B
// tiles stage into LDS as pure b128 copies (no in-GEMM transpose).
// ---------------------------------------------------------------------------

#define N_NODES 32768
#define E_EDGES 524288
#define F_ORI   32
#define HID     256
#define OUTD    16
#define MAXN    1024
#define BATCH   32
#define BN_EPS  1e-5f

typedef __attribute__((ext_vector_type(16))) __bf16           v16bf;
typedef __attribute__((ext_vector_type(16))) unsigned short   v16us;
typedef __attribute__((ext_vector_type(8)))  float            v8f;
typedef __attribute__((ext_vector_type(8)))  unsigned short   us8;

union FragU { v16us u; v16bf b; };

__device__ __forceinline__ unsigned short f2bf(float f) {
    unsigned int u = __float_as_uint(f);
    u += 0x7FFFu + ((u >> 16) & 1u);        // round-to-nearest-even
    return (unsigned short)(u >> 16);
}

// ---------------------------------------------------------------------------
// bf16 WMMA GEMM:  C[M,256] = A[M,K](bf16) @ Wt[256,K](bf16, pre-transposed)
//                  + bias (+ReLU)
// block = 256 threads (8 waves), tile 128x128, wave tile 32x64 (2x4 wmma).
// Global->LDS staging is register software-pipelined one K-tile ahead.
// ---------------------------------------------------------------------------
template <bool RELU>
__global__ __launch_bounds__(256, 2)
void k_gemm_wmma(const unsigned short* __restrict__ A,
                 const unsigned short* __restrict__ Wt,
                 const float* __restrict__ bias,
                 float* __restrict__ C,
                 int M, int K)
{
    __shared__ unsigned short As[128][40];   // +8 pad: rows 80B -> 16B aligned
    __shared__ unsigned short Bs[128][40];   // Bs[n][k], same layout as As

    const int tid   = threadIdx.x;
    const int lane  = tid & 31;
    const int wave  = tid >> 5;
    const int wm    = wave >> 1;             // 0..3  (M direction)
    const int wn    = wave & 1;              // 0..1  (N direction)
    const int row16 = lane & 15;
    const int kh    = lane >> 4;             // K-half select

    const int mblk = blockIdx.y * 128;
    const int nblk = blockIdx.x * 128;

    // staging assignment: thread -> (row r, 16-element half h) of the tile
    const int r = tid >> 1;                  // 0..127
    const int h = tid & 1;                   // 0..1
    const unsigned short* gA = A  + (size_t)(mblk + r) * K + h * 16;
    const unsigned short* gB = Wt + (size_t)(nblk + r) * K + h * 16;

    v8f acc[2][4];
    #pragma unroll
    for (int mi = 0; mi < 2; ++mi)
        #pragma unroll
        for (int ni = 0; ni < 4; ++ni)
            #pragma unroll
            for (int j = 0; j < 8; ++j)
                acc[mi][ni][j] = 0.0f;

    // prologue: fetch first K-tile into registers
    us8 ra0 = ((const us8*)gA)[0];
    us8 ra1 = ((const us8*)gA)[1];
    us8 rb0 = ((const us8*)gB)[0];
    us8 rb1 = ((const us8*)gB)[1];

    for (int k0 = 0; k0 < K; k0 += 32) {
        // ---- commit staged registers to LDS ----
        *(us8*)&As[r][h * 16]     = ra0;
        *(us8*)&As[r][h * 16 + 8] = ra1;
        *(us8*)&Bs[r][h * 16]     = rb0;
        *(us8*)&Bs[r][h * 16 + 8] = rb1;
        __syncthreads();

        // ---- issue next K-tile loads early (latency hidden behind WMMA) ----
        if (k0 + 32 < K) {
            const us8* nA = (const us8*)(gA + k0 + 32);
            const us8* nB = (const us8*)(gB + k0 + 32);
            ra0 = nA[0]; ra1 = nA[1];
            rb0 = nB[0]; rb1 = nB[1];
            if (k0 + 64 < K)
                __builtin_prefetch(gA + k0 + 64, 0, 0);
        }

        // ---- load fragments (2x ds_load_b128 each, per ISA bf16 layout) ----
        FragU a[2], b[4];
        #pragma unroll
        for (int mi = 0; mi < 2; ++mi) {
            const int rr = wm * 32 + mi * 16 + row16;
            us8 lo = *(const us8*)&As[rr][kh * 8];
            us8 hi = *(const us8*)&As[rr][16 + kh * 8];
            #pragma unroll
            for (int j = 0; j < 8; ++j) { a[mi].u[j] = lo[j]; a[mi].u[j + 8] = hi[j]; }
        }
        #pragma unroll
        for (int ni = 0; ni < 4; ++ni) {
            const int cc = wn * 64 + ni * 16 + row16;
            us8 lo = *(const us8*)&Bs[cc][kh * 8];
            us8 hi = *(const us8*)&Bs[cc][16 + kh * 8];
            #pragma unroll
            for (int j = 0; j < 8; ++j) { b[ni].u[j] = lo[j]; b[ni].u[j + 8] = hi[j]; }
        }

        // ---- 8 WMMAs per wave per K-step ----
        #pragma unroll
        for (int mi = 0; mi < 2; ++mi)
            #pragma unroll
            for (int ni = 0; ni < 4; ++ni)
                acc[mi][ni] = __builtin_amdgcn_wmma_f32_16x16x32_bf16(
                    false, a[mi].b, false, b[ni].b, (short)0, acc[mi][ni], false, false);

        __syncthreads();
    }

    // ---- epilogue: bias (+ReLU), store per 16x16 f32 C layout ----
    #pragma unroll
    for (int ni = 0; ni < 4; ++ni) {
        const int c  = nblk + wn * 64 + ni * 16 + row16;
        const float bv = bias[c];
        #pragma unroll
        for (int mi = 0; mi < 2; ++mi) {
            #pragma unroll
            for (int v = 0; v < 8; ++v) {
                const int rr = mblk + wm * 32 + mi * 16 + v + 8 * kh;
                float val = acc[mi][ni][v] + bv;
                if (RELU) val = fmaxf(val, 0.0f);
                C[(size_t)rr * HID + c] = val;
            }
        }
    }
}

// ---------------------------------------------------------------------------
// glue kernels
// ---------------------------------------------------------------------------
__global__ void k_concat(const float* __restrict__ xo, const float* __restrict__ g0,
                         const float* __restrict__ g1, const float* __restrict__ g2,
                         float* __restrict__ xf)
{
    const int n = blockIdx.x;
    const int j = threadIdx.x;                    // 0..127
    float v;
    if      (j < 32)  v = xo[(size_t)n * 32 + j];
    else if (j < 64)  v = g0[(size_t)n * 32 + (j - 32)];
    else if (j < 96)  v = g1[(size_t)n * 32 + (j - 64)];
    else              v = g2[(size_t)n * 32 + (j - 96)];
    xf[(size_t)n * 128 + j] = v;
}

__global__ void k_deg(const int* __restrict__ ei, float* __restrict__ deg)
{
    const int e = blockIdx.x * blockDim.x + threadIdx.x;
    if (e < E_EDGES) atomicAdd(&deg[ei[e]], 1.0f);   // row 0 = dst
}

__global__ void k_invdeg(const float* __restrict__ deg, float* __restrict__ invd)
{
    const int n = blockIdx.x * blockDim.x + threadIdx.x;
    if (n < N_NODES) {
        const float d = deg[n];
        invd[n] = (d > 0.0f) ? (1.0f / fmaxf(d, 1.0f)) : 0.0f;
    }
}

template <int HIN>
__global__ void k_scatter(const int* __restrict__ ei, const float* __restrict__ xf,
                          float* __restrict__ agg)
{
    const int e    = blockIdx.x * 8 + (threadIdx.x >> 5);
    const int lane = threadIdx.x & 31;
    if (e >= E_EDGES) return;
    const int dst = ei[e];
    const int src = ei[E_EDGES + e];
    #pragma unroll
    for (int f = lane; f < HIN; f += 32)
        atomicAdd(&agg[(size_t)dst * HIN + f], xf[(size_t)src * HIN + f]);
}

__global__ void k_prepA(const float* __restrict__ agg, const float* __restrict__ xf,
                        const float* __restrict__ invd, unsigned short* __restrict__ abf,
                        int Hin)
{
    const int n = blockIdx.y;
    const int k = blockIdx.x * 256 + threadIdx.x;
    const int K = Hin * 2;
    const float v = (k < Hin) ? agg[(size_t)n * Hin + k] * invd[n]
                              : xf[(size_t)n * Hin + (k - Hin)];
    abf[(size_t)n * K + k] = f2bf(v);
}

// build transposed bf16 weight Wt[c][r] = concat(Wl;Wr)[r][c], Wt is [256, K]
__global__ void k_prepWt(const float* __restrict__ Wl, const float* __restrict__ Wr,
                         unsigned short* __restrict__ wt, int Hin)
{
    const int c = blockIdx.y;                     // 0..255 (output column -> Wt row)
    const int rr = blockIdx.x * 256 + threadIdx.x; // 0..2*Hin-1
    const int K = Hin * 2;
    const float v = (rr < Hin) ? Wl[(size_t)rr * HID + c]
                               : Wr[(size_t)(rr - Hin) * HID + c];
    wt[(size_t)c * K + rr] = f2bf(v);
}

// transposed bf16 of a single [K,256] f32 weight -> Wt [256, K]
__global__ void k_prepWt1(const float* __restrict__ W, unsigned short* __restrict__ wt,
                          int K)
{
    const int c = blockIdx.y;                      // 0..255
    const int k = blockIdx.x * 256 + threadIdx.x;  // 0..K-1
    wt[(size_t)c * K + k] = f2bf(W[(size_t)k * HID + c]);
}

__global__ void k_f32bf16(const float* __restrict__ in, unsigned short* __restrict__ out,
                          int count)
{
    const int i = blockIdx.x * blockDim.x + threadIdx.x;
    if (i < count) out[i] = f2bf(in[i]);
}

__global__ void k_bnstats(const float* __restrict__ y, float* __restrict__ sum,
                          float* __restrict__ sumsq)
{
    const int c  = threadIdx.x;
    const int r0 = blockIdx.x * 128;
    float s = 0.0f, s2 = 0.0f;
    for (int r = 0; r < 128; ++r) {
        const float v = y[(size_t)(r0 + r) * HID + c];
        s += v; s2 += v * v;
    }
    atomicAdd(&sum[c], s);
    atomicAdd(&sumsq[c], s2);
}

__global__ void k_bnapply(const float* __restrict__ y, const float* __restrict__ sum,
                          const float* __restrict__ sumsq, const float* __restrict__ gamma,
                          const float* __restrict__ beta, float* __restrict__ xf)
{
    const int n = blockIdx.x;
    const int c = threadIdx.x;
    const float inv_n = 1.0f / (float)N_NODES;
    const float m   = sum[c] * inv_n;
    const float var = sumsq[c] * inv_n - m * m;
    const float sc  = rsqrtf(var + BN_EPS) * gamma[c];
    const float v   = (y[(size_t)n * HID + c] - m) * sc + beta[c];
    xf[(size_t)n * HID + c] = fmaxf(v, 0.0f);
}

__global__ void k_rowdot(const float* __restrict__ h, const float* __restrict__ w,
                         const float* __restrict__ b1b, float* __restrict__ z)
{
    const int r    = blockIdx.x * 8 + (threadIdx.x >> 5);
    const int lane = threadIdx.x & 31;
    float s = 0.0f;
    #pragma unroll
    for (int k = lane; k < HID; k += 32)
        s += h[(size_t)r * HID + k] * w[k];
    #pragma unroll
    for (int o = 16; o > 0; o >>= 1)
        s += __shfl_xor(s, o, 32);
    if (lane == 0) z[r] = s + b1b[0];
}

__global__ void k_bnfinal(const float* __restrict__ z, const float* __restrict__ gamma_f,
                          const float* __restrict__ beta_f, float* __restrict__ zn)
{
    const int c = threadIdx.x;
    float s = 0.0f, s2 = 0.0f;
    for (int b = 0; b < BATCH; ++b) {
        const float v = z[b * HID + c];
        s += v; s2 += v * v;
    }
    const float m   = s / (float)BATCH;
    const float var = s2 / (float)BATCH - m * m;
    const float sc  = rsqrtf(var + BN_EPS) * gamma_f[c];
    const float bt  = beta_f[c];
    for (int b = 0; b < BATCH; ++b) {
        const float v = (z[b * HID + c] - m) * sc + bt;
        zn[b * HID + c] = fmaxf(v, 0.0f);
    }
}

__global__ void k_mlp2(const float* __restrict__ zn, const float* __restrict__ W2a,
                       const float* __restrict__ b2a, const float* __restrict__ W2b,
                       const float* __restrict__ b2b, float* __restrict__ out)
{
    __shared__ float zr[HID];
    __shared__ float t[HID];
    const int b = blockIdx.x;
    const int c = threadIdx.x;
    zr[c] = zn[b * HID + c];
    __syncthreads();
    float s = b2a[c];
    for (int k = 0; k < HID; ++k) s += zr[k] * W2a[(size_t)k * HID + c];
    t[c] = fmaxf(s, 0.0f);
    __syncthreads();
    if (c < OUTD) {
        float o = b2b[c];
        for (int k = 0; k < HID; ++k) o += t[k] * W2b[(size_t)k * OUTD + c];
        out[b * OUTD + c] = o;
    }
}

// ---------------------------------------------------------------------------
// launch
// ---------------------------------------------------------------------------
extern "C" void kernel_launch(void* const* d_in, const int* in_sizes, int n_in,
                              void* d_out, int out_size, void* d_ws, size_t ws_size,
                              hipStream_t stream)
{
    (void)in_sizes; (void)n_in; (void)out_size; (void)ws_size;

    const float* x_ori   = (const float*)d_in[0];
    const float* g0      = (const float*)d_in[1];
    const float* g1      = (const float*)d_in[2];
    const float* g2      = (const float*)d_in[3];
    const int*   ei      = (const int*)  d_in[4];
    const float* Wl0     = (const float*)d_in[5];
    const float* Wr0     = (const float*)d_in[6];
    const float* bb0     = (const float*)d_in[7];
    const float* Wl      = (const float*)d_in[8];
    const float* Wr      = (const float*)d_in[9];
    const float* bb      = (const float*)d_in[10];
    const float* gamma   = (const float*)d_in[11];
    const float* beta    = (const float*)d_in[12];
    const float* W1a     = (const float*)d_in[13];
    const float* b1a     = (const float*)d_in[14];
    const float* W1b     = (const float*)d_in[15];
    const float* b1b     = (const float*)d_in[16];
    const float* gamma_f = (const float*)d_in[17];
    const float* beta_f  = (const float*)d_in[18];
    const float* W2a     = (const float*)d_in[19];
    const float* b2a     = (const float*)d_in[20];
    const float* W2b     = (const float*)d_in[21];
    const float* b2b     = (const float*)d_in[22];
    float* out = (float*)d_out;

    // workspace carve-up (~101 MB)
    float* xf   = (float*)d_ws;                                   // N*256 f32
    float* yagg = xf + (size_t)N_NODES * HID;                     // N*256 f32 (agg, then y, then h)
    unsigned short* abf = (unsigned short*)(yagg + (size_t)N_NODES * HID); // N*512 bf16
    unsigned short* wbf = abf + (size_t)N_NODES * 512;            // 256*1024 bf16 (transposed)
    float* deg   = (float*)(wbf + (size_t)HID * 1024);            // N
    float* invd  = deg + N_NODES;                                 // N
    float* bnsum = invd + N_NODES;                                // 256
    float* bnsq  = bnsum + HID;                                   // 256
    float* zvec  = bnsq + HID;                                    // 8192
    float* zn    = zvec + BATCH * HID;                            // 8192

    // ---- input prep ----
    k_concat<<<N_NODES, 128, 0, stream>>>(x_ori, g0, g1, g2, xf);
    hipMemsetAsync(deg, 0, N_NODES * sizeof(float), stream);
    k_deg<<<E_EDGES / 256, 256, 0, stream>>>(ei, deg);
    k_invdeg<<<N_NODES / 256, 256, 0, stream>>>(deg, invd);

    // ---- 4 SAGE layers ----
    for (int l = 0; l < 4; ++l) {
        const int Hin = (l == 0) ? 128 : HID;
        const int K   = 2 * Hin;
        const float* wl = (l == 0) ? Wl0 : (Wl + (size_t)(l - 1) * HID * HID);
        const float* wr = (l == 0) ? Wr0 : (Wr + (size_t)(l - 1) * HID * HID);
        const float* bl = (l == 0) ? bb0 : (bb + (size_t)(l - 1) * HID);

        hipMemsetAsync(yagg, 0, (size_t)N_NODES * Hin * sizeof(float), stream);
        if (Hin == 128)
            k_scatter<128><<<E_EDGES / 8, 256, 0, stream>>>(ei, xf, yagg);
        else
            k_scatter<256><<<E_EDGES / 8, 256, 0, stream>>>(ei, xf, yagg);

        k_prepA<<<dim3(K / 256, N_NODES), 256, 0, stream>>>(yagg, xf, invd, abf, Hin);
        k_prepWt<<<dim3(K / 256, HID), 256, 0, stream>>>(wl, wr, wbf, Hin);

        k_gemm_wmma<false><<<dim3(HID / 128, N_NODES / 128), 256, 0, stream>>>(
            abf, wbf, bl, yagg, N_NODES, K);

        hipMemsetAsync(bnsum, 0, 2 * HID * sizeof(float), stream);
        k_bnstats<<<N_NODES / 128, 256, 0, stream>>>(yagg, bnsum, bnsq);
        k_bnapply<<<N_NODES, 256, 0, stream>>>(yagg, bnsum, bnsq,
                                               gamma + l * HID, beta + l * HID, xf);
    }

    // ---- mlp1: [8192,1024] @ [1024,256] + b1a, ReLU ----
    k_f32bf16<<<(N_NODES * HID) / 256, 256, 0, stream>>>(xf, abf, N_NODES * HID);
    k_prepWt1<<<dim3(MAXN / 256, HID), 256, 0, stream>>>(W1a, wbf, MAXN);
    k_gemm_wmma<true><<<dim3(HID / 128, (BATCH * HID) / 128), 256, 0, stream>>>(
        abf, wbf, b1a, yagg, BATCH * HID, MAXN);

    // ---- h @ W1b + b1b -> z[32,256] ----
    k_rowdot<<<(BATCH * HID) / 8, 256, 0, stream>>>(yagg, W1b, b1b, zvec);

    // ---- final BN (over 32 rows) + ReLU ----
    k_bnfinal<<<1, 256, 0, stream>>>(zvec, gamma_f, beta_f, zn);

    // ---- mlp2 -> out[32,16] ----
    k_mlp2<<<BATCH, 256, 0, stream>>>(zn, W2a, b2a, W2b, b2b, out);
}
